// MaterialEdgePredictor_50362786513102
// MI455X (gfx1250) — matrix operand order repaired and checked by reading, not compiled, not run on hardware
//
#include <hip/hip_runtime.h>
#include <hip/hip_bf16.h>
#include <math.h>

// MI455X / gfx1250: wave32, WMMA f32<-f16 16x16x32 path.
// Weights are pre-packed (f32 -> f16, WMMA fragment order) once per call,
// then staged block-wide into LDS; inner GEMM loop reads B via ds_load_b128.

typedef __attribute__((ext_vector_type(16))) _Float16 v16h;
typedef __attribute__((ext_vector_type(8)))  float    v8f;

#define HD 128   // hidden channels

// 16-bit A-matrix 16x32 element->k mapping (ISA 7.12.2):
// lanes 0-15: e0..7 -> k0..7, e8..15 -> k16..23 ; lanes 16-31: +8
__device__ __forceinline__ int a_koff(int e, int hi) {
  return (e & 7) + ((e >> 3) << 4) + (hi ? 8 : 0);
}
// B-matrix 32x16: same k mapping as A (lanes 0-15: e0..7 -> k0..7, e8..15 -> k16..23 ; lanes 16-31: +8)
__device__ __forceinline__ int b_koff(int e, int hi) {
  return (e & 7) + ((e >> 3) << 4) + (hi ? 8 : 0);
}

// ---------------------------------------------------------------------------
// Pack B_virtual [K x N] = [B0 ; B1] (split at K0) into f16 WMMA fragment
// order: element idx = ((ks*(N/16) + ct)*32 + lane)*16 + e.
// ---------------------------------------------------------------------------
__global__ __launch_bounds__(256)
void pack_b_kernel(const float* __restrict__ B0, const float* __restrict__ B1,
                   int K0, _Float16* __restrict__ out, int K, int N)
{
  const int idx = blockIdx.x * 256 + threadIdx.x;
  if (idx >= K * N) return;
  const int e    = idx & 15;
  const int lane = (idx >> 4) & 31;
  const int f    = idx >> 9;
  const int CT   = N / 16;
  const int ks   = f / CT;
  const int ct   = f - ks * CT;
  const int hi   = lane >> 4;
  const int n    = ct * 16 + (lane & 15);
  const int k    = ks * 32 + b_koff(e, hi);
  const float v  = (k < K0) ? B0[k * N + n] : B1[(k - K0) * N + n];
  out[idx] = (_Float16)v;
}

// ---------------------------------------------------------------------------
// Generic WMMA GEMM: out[M, 128] = act( [A0*scale | A1] @ Bp + bias )
// Bp is pre-packed f16 fragments for the full K x 128 virtual B.
// ---------------------------------------------------------------------------
template<int KSTEPS, int K0STEPS, bool RELU>
__global__ __launch_bounds__(256)
void wmma_gemm_kernel(const float* __restrict__ A0,
                      const float* __restrict__ A1,
                      const float* __restrict__ cnt,
                      const _Float16* __restrict__ Bp,
                      const float* __restrict__ bias,
                      float* __restrict__ out,
                      int M)
{
  constexpr int K0 = K0STEPS * 32;
  constexpr int K1 = KSTEPS * 32 - K0;
  constexpr int CT = HD / 16;
  constexpr int NFRAG = KSTEPS * CT * 32;        // # v16h fragments

  extern __shared__ __align__(16) char smem_raw[];
  _Float16* bsh = (_Float16*)smem_raw;

  // Block-wide stage of packed weights into LDS (b128 copies).
  {
    const uint4* gp = (const uint4*)Bp;
    uint4* sp = (uint4*)bsh;
    #pragma unroll 1
    for (int i = threadIdx.x; i < NFRAG * 2; i += 256)
      sp[i] = gp[i];
  }

  const int lane = threadIdx.x & 31;
  const int wave = threadIdx.x >> 5;
  const int hi   = lane >> 4;
  const int l15  = lane & 15;
  const int rowbase = blockIdx.x * 128 + wave * 16;

  int arow = rowbase + l15;
  if (arow >= M) arow = M - 1;           // clamp for fragment loads

  float rscale = 1.0f;
  if (cnt) rscale = 1.0f / fmaxf(cnt[arow], 1.0f);

  const float* a0p = A0 + (size_t)arow * K0;
  const float* a1p = (K1 > 0) ? (A1 + (size_t)arow * (size_t)K1) : nullptr;

  // Preload all A fragments into registers.
  v16h afrag[KSTEPS];
  #pragma unroll
  for (int ks = 0; ks < KSTEPS; ++ks) {
    #pragma unroll
    for (int e = 0; e < 16; ++e) {
      const int k = ks * 32 + a_koff(e, hi);
      float v;
      if (K1 == 0)      v = a0p[k] * rscale;
      else if (k < K0)  v = a0p[k] * rscale;
      else              v = a1p[k - K0];
      afrag[ks][e] = (_Float16)v;
    }
  }

  __syncthreads();
  const v16h* bfr = (const v16h*)bsh;

  #pragma unroll 1
  for (int ct = 0; ct < CT; ++ct) {
    const int n = ct * 16 + l15;
    v8f acc = {};
    #pragma unroll
    for (int ks = 0; ks < KSTEPS; ++ks) {
      const v16h bfrag = bfr[(ks * CT + ct) * 32 + lane];
      acc = __builtin_amdgcn_wmma_f32_16x16x32_f16(
          false, afrag[ks], false, bfrag, (short)0, acc, false, false);
    }
    const float bv = bias ? bias[n] : 0.0f;
    #pragma unroll
    for (int v = 0; v < 8; ++v) {
      const int orow = rowbase + v + hi * 8;   // D layout: vgpr=row, lane=col
      if (orow < M) {
        float x = acc[v] + bv;
        if (RELU) x = fmaxf(x, 0.0f);
        out[(size_t)orow * HD + n] = x;
      }
    }
  }
}

// ---------------------------------------------------------------------------
// Edge decoder, fully fused: gather concat(h_src, h_dst) -> relu(z@Wmlp+b)
// -> (.@Wout+b) -> sigmoid. 256->1 layer folded into the WMMA epilogue.
// ---------------------------------------------------------------------------
__global__ __launch_bounds__(256)
void decoder_kernel(const float* __restrict__ hs, const float* __restrict__ hd,
                    const int* __restrict__ isrc, const int* __restrict__ idst,
                    const _Float16* __restrict__ pMlp,
                    const float* __restrict__ bmlp,
                    const float* __restrict__ Wout, const float* __restrict__ bout,
                    float* __restrict__ out, int M)
{
  constexpr int KSTEPS = 8;    // K = 256 = [h_src | h_dst]
  constexpr int CT = 16;       // N = 256
  constexpr int NFRAG = KSTEPS * CT * 32;

  extern __shared__ __align__(16) char smem_raw[];
  _Float16* bsh = (_Float16*)smem_raw;
  {
    const uint4* gp = (const uint4*)pMlp;
    uint4* sp = (uint4*)bsh;
    #pragma unroll 1
    for (int i = threadIdx.x; i < NFRAG * 2; i += 256)
      sp[i] = gp[i];
  }

  const int lane = threadIdx.x & 31;
  const int wave = threadIdx.x >> 5;
  const int hi   = lane >> 4;
  const int l15  = lane & 15;
  const int rowbase = blockIdx.x * 128 + wave * 16;

  int arow = rowbase + l15;
  if (arow >= M) arow = M - 1;
  const float* sp = hs + (size_t)isrc[arow] * HD;
  const float* dp = hd + (size_t)idst[arow] * HD;

  v16h afrag[KSTEPS];
  #pragma unroll
  for (int ks = 0; ks < KSTEPS; ++ks) {
    #pragma unroll
    for (int e = 0; e < 16; ++e) {
      const int k = ks * 32 + a_koff(e, hi);
      afrag[ks][e] = (_Float16)((k < HD) ? sp[k] : dp[k - HD]);
    }
  }

  __syncthreads();
  const v16h* bfr = (const v16h*)bsh;

  float partial[8] = {0.f, 0.f, 0.f, 0.f, 0.f, 0.f, 0.f, 0.f};
  #pragma unroll 1
  for (int ct = 0; ct < CT; ++ct) {
    const int n = ct * 16 + l15;
    v8f acc = {};
    #pragma unroll
    for (int ks = 0; ks < KSTEPS; ++ks) {
      const v16h bfrag = bfr[(ks * CT + ct) * 32 + lane];
      acc = __builtin_amdgcn_wmma_f32_16x16x32_f16(
          false, afrag[ks], false, bfrag, (short)0, acc, false, false);
    }
    const float bv = bmlp[n];
    const float wv = Wout[n];
    #pragma unroll
    for (int v = 0; v < 8; ++v)
      partial[v] += fmaxf(acc[v] + bv, 0.0f) * wv;   // relu + out-layer dot
  }

  #pragma unroll
  for (int off = 1; off <= 8; off <<= 1) {
    #pragma unroll
    for (int v = 0; v < 8; ++v)
      partial[v] += __shfl_xor(partial[v], off, 32);
  }
  if (l15 < 8) {
    float x = partial[0];
    #pragma unroll
    for (int v = 1; v < 8; ++v)
      if (l15 == v) x = partial[v];
    const int orow = rowbase + l15 + hi * 8;
    if (orow < M) {
      x += bout[0];
      out[orow] = 1.0f / (1.0f + __expf(-x));
    }
  }
}

// Segment-sum via f32 atomics: 32 threads per edge (4 floats each).
__global__ __launch_bounds__(256)
void scatter_mean_kernel(const float* __restrict__ z,
                         const int* __restrict__ src, const int* __restrict__ dst,
                         float* __restrict__ agg, float* __restrict__ cnt,
                         int nedges)
{
  const long long t = (long long)blockIdx.x * blockDim.x + threadIdx.x;
  const int edge = (int)(t >> 5);
  const int q    = (int)(t & 31);
  if (edge >= nedges) return;
  const int s = src[edge];
  const int d = dst[edge];
  const float4 v = *(const float4*)(z + (size_t)s * HD + q * 4);
  float* a = agg + (size_t)d * HD + q * 4;
  atomicAdd(a + 0, v.x);
  atomicAdd(a + 1, v.y);
  atomicAdd(a + 2, v.z);
  atomicAdd(a + 3, v.w);
  if (q == 0) atomicAdd(cnt + d, 1.0f);
}

__global__ __launch_bounds__(256)
void zero_f32_kernel(float* __restrict__ p, long long n)
{
  const long long i = ((long long)blockIdx.x * blockDim.x + threadIdx.x) * 4;
  if (i + 3 < n) {
    *(float4*)(p + i) = make_float4(0.f, 0.f, 0.f, 0.f);
  } else {
    for (int j = 0; j < 4; ++j)
      if (i + j < n) p[i + j] = 0.f;
  }
}

extern "C" void kernel_launch(void* const* d_in, const int* in_sizes, int n_in,
                              void* d_out, int out_size, void* d_ws, size_t ws_size,
                              hipStream_t stream)
{
  (void)n_in; (void)out_size; (void)ws_size;

  const float* x_m     = (const float*)d_in[0];
  const float* x_e     = (const float*)d_in[1];
  const int*   src_m2e = (const int*)d_in[2];
  const int*   dst_m2e = (const int*)d_in[3];
  const int*   src_e2m = (const int*)d_in[4];
  const int*   dst_e2m = (const int*)d_in[5];
  const int*   ls_m2e  = (const int*)d_in[6];
  const int*   ld_m2e  = (const int*)d_in[7];
  const int*   ls_e2m  = (const int*)d_in[8];
  const int*   ld_e2m  = (const int*)d_in[9];
  const float* Wp_m    = (const float*)d_in[10]; const float* bp_m    = (const float*)d_in[11];
  const float* Wp_e    = (const float*)d_in[12]; const float* bp_e    = (const float*)d_in[13];
  const float* Wl_m2e  = (const float*)d_in[14]; const float* bl_m2e  = (const float*)d_in[15];
  const float* Wr_m2e  = (const float*)d_in[16];
  const float* Wl_e2m  = (const float*)d_in[17]; const float* bl_e2m  = (const float*)d_in[18];
  const float* Wr_e2m  = (const float*)d_in[19];
  const float* Wffw_m  = (const float*)d_in[20]; const float* bffw_m  = (const float*)d_in[21];
  const float* Wffw_e  = (const float*)d_in[22]; const float* bffw_e  = (const float*)d_in[23];
  const float* Wproj_m = (const float*)d_in[24]; const float* bproj_m = (const float*)d_in[25];
  const float* Wproj_e = (const float*)d_in[26]; const float* bproj_e = (const float*)d_in[27];
  const float* Wmlp_m2e = (const float*)d_in[28]; const float* bmlp_m2e = (const float*)d_in[29];
  const float* Wout_m2e = (const float*)d_in[30]; const float* bout_m2e = (const float*)d_in[31];
  const float* Wmlp_e2m = (const float*)d_in[32]; const float* bmlp_e2m = (const float*)d_in[33];
  const float* Wout_e2m = (const float*)d_in[34]; const float* bout_e2m = (const float*)d_in[35];

  const int NM = in_sizes[0] / 256;   // 100000 materials
  const int NE = in_sizes[1] / 64;    // 50000 elements
  const int E  = in_sizes[2];         // 1M message edges per direction
  const int EL = in_sizes[6];         // 250K label edges per decoder

  // ---- workspace layout: f32 feature buffers, then packed f16 weights ----
  float* ws = (float*)d_ws;
  size_t o = 0;
  float* zM   = ws + o; o += (size_t)NM * HD;
  float* zE   = ws + o; o += (size_t)NE * HD;
  float* hM   = ws + o; o += (size_t)NM * HD;
  float* hE   = ws + o; o += (size_t)NE * HD;
  float* aggM = ws + o; o += (size_t)NM * HD;
  float* aggE = ws + o; o += (size_t)NE * HD;
  float* cntM = ws + o; o += (size_t)NM;
  float* cntE = ws + o; o += (size_t)NE;

  _Float16* pk = (_Float16*)(ws + o);
  size_t po = 0;
  _Float16* pWp_m   = pk + po; po += 256 * 128;
  _Float16* pWp_e   = pk + po; po += 64 * 128;
  _Float16* pSage_m = pk + po; po += 256 * 128;   // [Wl_e2m ; Wr_e2m]
  _Float16* pSage_e = pk + po; po += 256 * 128;   // [Wl_m2e ; Wr_m2e]
  _Float16* pFfw_m  = pk + po; po += 128 * 128;
  _Float16* pFfw_e  = pk + po; po += 128 * 128;
  _Float16* pProj_m = pk + po; po += 128 * 128;
  _Float16* pProj_e = pk + po; po += 128 * 128;
  _Float16* pMlp_m2e = pk + po; po += 256 * 256;
  _Float16* pMlp_e2m = pk + po; po += 256 * 256;

  const dim3 blk(256);
  auto gb = [](long long rows) { return (unsigned)((rows + 127) / 128); };
  auto pb = [](int K, int N) { return (unsigned)(((long long)K * N + 255) / 256); };

  // 0) pack all weights to f16 fragment order (tiny, one-time per call)
  pack_b_kernel<<<pb(256, 128), blk, 0, stream>>>(Wp_m, nullptr, 256, pWp_m, 256, 128);
  pack_b_kernel<<<pb(64, 128),  blk, 0, stream>>>(Wp_e, nullptr, 64,  pWp_e, 64, 128);
  pack_b_kernel<<<pb(256, 128), blk, 0, stream>>>(Wl_e2m, Wr_e2m, 128, pSage_m, 256, 128);
  pack_b_kernel<<<pb(256, 128), blk, 0, stream>>>(Wl_m2e, Wr_m2e, 128, pSage_e, 256, 128);
  pack_b_kernel<<<pb(128, 128), blk, 0, stream>>>(Wffw_m, nullptr, 128, pFfw_m, 128, 128);
  pack_b_kernel<<<pb(128, 128), blk, 0, stream>>>(Wffw_e, nullptr, 128, pFfw_e, 128, 128);
  pack_b_kernel<<<pb(128, 128), blk, 0, stream>>>(Wproj_m, nullptr, 128, pProj_m, 128, 128);
  pack_b_kernel<<<pb(128, 128), blk, 0, stream>>>(Wproj_e, nullptr, 128, pProj_e, 128, 128);
  pack_b_kernel<<<pb(256, 256), blk, 0, stream>>>(Wmlp_m2e, nullptr, 256, pMlp_m2e, 256, 256);
  pack_b_kernel<<<pb(256, 256), blk, 0, stream>>>(Wmlp_e2m, nullptr, 256, pMlp_e2m, 256, 256);

  // 1) zero agg + cnt (contiguous region)
  const long long nzero = (long long)NM * HD + (long long)NE * HD + NM + NE;
  zero_f32_kernel<<<(unsigned)((nzero / 4 + 255) / 256), blk, 0, stream>>>(aggM, nzero);

  // LDS bytes for each KSTEPS (KSTEPS * 8 coltiles * 32 lanes * 32B)
  const size_t lds8 = 8 * 8 * 32 * 32;   // 64 KB
  const size_t lds4 = 4 * 8 * 32 * 32;   // 32 KB
  const size_t lds2 = 2 * 8 * 32 * 32;   // 16 KB
  const size_t ldsD = 8 * 16 * 32 * 32;  // 128 KB (within 320KB WGP LDS)

  // 2) input projections: z = x @ Wp + bp
  wmma_gemm_kernel<8, 8, false><<<gb(NM), blk, lds8, stream>>>(
      x_m, nullptr, nullptr, pWp_m, bp_m, zM, NM);
  wmma_gemm_kernel<2, 2, false><<<gb(NE), blk, lds2, stream>>>(
      x_e, nullptr, nullptr, pWp_e, bp_e, zE, NE);

  // 3) edge scatter-add (segment sum + counts)
  const unsigned sblk = (unsigned)(((long long)E * 32 + 255) / 256);
  scatter_mean_kernel<<<sblk, blk, 0, stream>>>(zE, src_e2m, dst_e2m, aggM, cntM, E);
  scatter_mean_kernel<<<sblk, blk, 0, stream>>>(zM, src_m2e, dst_m2e, aggE, cntE, E);

  // 4) fused SAGE: h = relu([mean | root] @ [Wl ; Wr] + bl)
  wmma_gemm_kernel<8, 4, true><<<gb(NM), blk, lds8, stream>>>(
      aggM, zM, cntM, pSage_m, bl_e2m, hM, NM);
  wmma_gemm_kernel<8, 4, true><<<gb(NE), blk, lds8, stream>>>(
      aggE, zE, cntE, pSage_e, bl_m2e, hE, NE);

  // 5) ffw (relu) — reuse z buffers
  wmma_gemm_kernel<4, 4, true><<<gb(NM), blk, lds4, stream>>>(
      hM, nullptr, nullptr, pFfw_m, bffw_m, zM, NM);
  wmma_gemm_kernel<4, 4, true><<<gb(NE), blk, lds4, stream>>>(
      hE, nullptr, nullptr, pFfw_e, bffw_e, zE, NE);

  // 6) output projection — back into h buffers
  wmma_gemm_kernel<4, 4, false><<<gb(NM), blk, lds4, stream>>>(
      zM, nullptr, nullptr, pProj_m, bproj_m, hM, NM);
  wmma_gemm_kernel<4, 4, false><<<gb(NE), blk, lds4, stream>>>(
      zE, nullptr, nullptr, pProj_e, bproj_e, hE, NE);

  // 7) fused edge decoders -> d_out = [out_m2e (EL) | out_e2m (EL)]
  float* outp = (float*)d_out;
  decoder_kernel<<<gb(EL), blk, ldsD, stream>>>(
      hM, hE, ls_m2e, ld_m2e, pMlp_m2e, bmlp_m2e, Wout_m2e, bout_m2e, outp, EL);
  decoder_kernel<<<gb(EL), blk, ldsD, stream>>>(
      hE, hM, ls_e2m, ld_e2m, pMlp_e2m, bmlp_e2m, Wout_e2m, bout_e2m, outp + EL, EL);
}
